// SelfAttention_83330955476996
// MI455X (gfx1250) — matrix operand order repaired and checked by reading, not compiled, not run on hardware
//
#include <hip/hip_runtime.h>

typedef __attribute__((ext_vector_type(16))) _Float16 v16h;
typedef __attribute__((ext_vector_type(8)))  float    v8f;
typedef int v4i __attribute__((vector_size(16)));

#define CDIM 128
#define NDIM 4096   // 64*64
#define BDIM 8

union Frag16 { v16h v; uint4 u[2]; };

__device__ __forceinline__ v8f wmma_f16(const Frag16& a, const Frag16& b, v8f c) {
  return __builtin_amdgcn_wmma_f32_16x16x32_f16(false, a.v, false, b.v, (short)0, c, false, false);
}

// ---- async global->LDS copy (ASYNCcnt path), with sync fallback ----
#if __has_builtin(__builtin_amdgcn_global_load_async_to_lds_b128)
#define HAVE_ASYNC 1
typedef __attribute__((address_space(1))) v4i g_v4i;
typedef __attribute__((address_space(3))) v4i l_v4i;
__device__ __forceinline__ void async_cp16(void* lds, const void* g) {
  __builtin_amdgcn_global_load_async_to_lds_b128((g_v4i*)g, (l_v4i*)lds, 0, 0);
}
#else
#define HAVE_ASYNC 0
__device__ __forceinline__ void async_cp16(void* lds, const void* g) {
  *(uint4*)lds = *(const uint4*)g;
}
#endif

__device__ __forceinline__ void wait_async() {
#if __has_builtin(__builtin_amdgcn_s_wait_asynccnt)
  __builtin_amdgcn_s_wait_asynccnt(0);
#elif HAVE_ASYNC
  asm volatile("s_wait_asynccnt 0x0" ::: "memory");
#endif
}

// ---------------- kernel 1: w_out f32 -> f16 ----------------
__global__ void wconv_kernel(const float* __restrict__ w, _Float16* __restrict__ wh, int n) {
  int i = blockIdx.x * blockDim.x + threadIdx.x;
  if (i < n) wh[i] = (_Float16)w[i];
}

// ---------------- kernel 2: QKV projection ----------------
// qkv[b][o][nn] = sum_c w_qkv[o][c] * x[b][c][nn]
// Q,K stored token-major [b][n][c] (f16), V stored channel-major [b][c][n] (f16)
__global__ void qkv_kernel(const float* __restrict__ x, const float* __restrict__ wqkv,
                           _Float16* __restrict__ Qh, _Float16* __restrict__ Kh,
                           _Float16* __restrict__ Vh) {
  int idx = blockIdx.x * 256 + threadIdx.x;
  int nn = idx & (NDIM - 1);
  int o  = (idx >> 12) % 384;
  int b  = idx / (NDIM * 384);
  const float* wr = wqkv + o * CDIM;
  const float* xc = x + (size_t)b * CDIM * NDIM + nn;
  float acc = 0.f;
#pragma unroll 8
  for (int c = 0; c < CDIM; ++c) acc = fmaf(wr[c], xc[(size_t)c * NDIM], acc);
  _Float16 h = (_Float16)acc;
  if (o < CDIM)            Qh[((size_t)b * NDIM + nn) * CDIM + o] = h;
  else if (o < 2 * CDIM)   Kh[((size_t)b * NDIM + nn) * CDIM + (o - CDIM)] = h;
  else                     Vh[((size_t)b * CDIM + (o - 2 * CDIM)) * NDIM + nn] = h;
}

// ---------------- kernel 3: flash attention ----------------
// One wave owns a 16-row query tile (16 x 128 output). 8 waves/block.
// Double-buffered async K/V staging; per 32-key chunk: 8 S-WMMAs + softmax + 8 PV-WMMAs.
__global__ __launch_bounds__(256) void attn_kernel(const _Float16* __restrict__ Qh,
                                                   const _Float16* __restrict__ Kh,
                                                   const _Float16* __restrict__ Vh,
                                                   _Float16* __restrict__ Oh) {
  __shared__ __align__(16) _Float16 Ks[2][32 * CDIM];   // [buf][key][c]
  __shared__ __align__(16) _Float16 Vs[2][CDIM * 32];   // [buf][c][key]
  __shared__ __align__(16) _Float16 Pb[8][16 * 32];     // per-wave P tile [row][key]

  const int tid  = threadIdx.x;
  const int wave = tid >> 5;
  const int lane = tid & 31;
  const int b    = blockIdx.y;
  const int i0   = (blockIdx.x * 8 + wave) * 16;
  const int lrow = lane & 15;
  const int lhi  = lane >> 4;

  // Load Q fragments (A operand): 4 chunks of K=32 channels
  Frag16 qf[4];
  {
    const _Float16* qbase = Qh + ((size_t)b * NDIM + i0 + lrow) * CDIM;
#pragma unroll
    for (int kc = 0; kc < 4; ++kc) {
      int lo = kc * 32 + (lhi ? 8 : 0);
      qf[kc].u[0] = *(const uint4*)(qbase + lo);
      qf[kc].u[1] = *(const uint4*)(qbase + lo + 16);
    }
  }

  v8f o_acc[8] = {};
  float m_r[8], l_r[8];
#pragma unroll
  for (int r = 0; r < 8; ++r) { m_r[r] = -__builtin_inff(); l_r[r] = 0.f; }

  const _Float16* KgB = Kh + (size_t)b * NDIM * CDIM;
  const _Float16* VgB = Vh + (size_t)b * CDIM * NDIM;

  const int krow = tid >> 3, kseg = tid & 7;   // K staging: 32 rows x 256B
  const int vc   = tid >> 1, vh   = tid & 1;   // V staging: 128 rows x 64B

  auto stage = [&](int j0, int bufi) {
    _Float16* kd = &Ks[bufi][krow * CDIM + kseg * 16];
    const _Float16* ks = KgB + (size_t)(j0 + krow) * CDIM + kseg * 16;
    async_cp16(kd, ks);
    async_cp16(kd + 8, ks + 8);
    _Float16* vd = &Vs[bufi][vc * 32 + vh * 16];
    const _Float16* vsrc = VgB + (size_t)vc * NDIM + j0 + vh * 16;
    async_cp16(vd, vsrc);
    async_cp16(vd + 8, vsrc + 8);
  };

  stage(0, 0);

  for (int it = 0; it < NDIM / 32; ++it) {
    const int j0 = it * 32;
    const _Float16* ksb = &Ks[it & 1][0];
    const _Float16* vsb = &Vs[it & 1][0];

    wait_async();
    __syncthreads();                       // buf[it&1] fully staged by all threads
    if (j0 + 32 < NDIM) stage(j0 + 32, (it + 1) & 1);  // overlap DMA with compute

    // S tiles for keys [j0, j0+16) and [j0+16, j0+32)
    v8f s0 = {}; v8f s1 = {};
#pragma unroll
    for (int kc = 0; kc < 4; ++kc) {
      int cb = kc * 32 + lhi * 8;
      Frag16 kb0, kb1;
      const uint4* k0 = (const uint4*)(ksb + lrow * CDIM + cb);
      kb0.u[0] = k0[0]; kb0.u[1] = k0[2];
      const uint4* k1 = (const uint4*)(ksb + (lrow + 16) * CDIM + cb);
      kb1.u[0] = k1[0]; kb1.u[1] = k1[2];
      s0 = wmma_f16(qf[kc], kb0, s0);
      s1 = wmma_f16(qf[kc], kb1, s1);
    }

    // online softmax (row = C-layout VGPR index; reduce across 16-lane halves)
    float p0[8], p1[8], osc[8];
#pragma unroll
    for (int r = 0; r < 8; ++r) {
      float mx = fmaxf(s0[r], s1[r]);
#pragma unroll
      for (int msk = 1; msk <= 8; msk <<= 1) mx = fmaxf(mx, __shfl_xor(mx, msk, 32));
      float mn = fmaxf(m_r[r], mx);
      float sc = __expf(m_r[r] - mn);
      float e0 = __expf(s0[r] - mn);
      float e1 = __expf(s1[r] - mn);
      float rs = e0 + e1;
#pragma unroll
      for (int msk = 1; msk <= 8; msk <<= 1) rs += __shfl_xor(rs, msk, 32);
      l_r[r] = l_r[r] * sc + rs;
      m_r[r] = mn;
      osc[r] = sc;
      p0[r] = e0; p1[r] = e1;
    }
#pragma unroll
    for (int nc = 0; nc < 8; ++nc)
#pragma unroll
      for (int r = 0; r < 8; ++r) o_acc[nc][r] = o_acc[nc][r] * osc[r];

    // P: C-layout -> row-major in LDS, then reload as A fragment
    {
      _Float16* pw = &Pb[wave][0];
      int rb = lhi * 8;
#pragma unroll
      for (int r = 0; r < 8; ++r) {
        pw[(rb + r) * 32 + lrow]      = (_Float16)p0[r];
        pw[(rb + r) * 32 + lrow + 16] = (_Float16)p1[r];
      }
    }
    Frag16 pf;
    {
      const _Float16* pr = &Pb[wave][0] + lrow * 32 + (lhi ? 8 : 0);
      pf.u[0] = *(const uint4*)(pr);
      pf.u[1] = *(const uint4*)(pr + 16);
    }

    // O += P * V over 8 channel chunks of 16
#pragma unroll
    for (int nc = 0; nc < 8; ++nc) {
      Frag16 vb;
      const uint4* vp = (const uint4*)(vsb + (nc * 16 + lrow) * 32 + lhi * 8);
      vb.u[0] = vp[0]; vb.u[1] = vp[2];
      o_acc[nc] = wmma_f16(pf, vb, o_acc[nc]);
    }

    __syncthreads();   // all reads of buf[it&1] done before it is re-staged at it+2
  }

  // normalize and store O token-major [b][n][c] as f16
#pragma unroll
  for (int r = 0; r < 8; ++r) l_r[r] = 1.f / l_r[r];
#pragma unroll
  for (int r = 0; r < 8; ++r) {
    _Float16* orow = Oh + ((size_t)b * NDIM + i0 + lhi * 8 + r) * CDIM + lrow;
#pragma unroll
    for (int nc = 0; nc < 8; ++nc)
      orow[nc * 16] = (_Float16)(o_acc[nc][r] * l_r[r]);
  }
}

// ---------------- kernel 4: output projection + bias + residual ----------------
__global__ __launch_bounds__(256) void proj_kernel(const _Float16* __restrict__ Wh,
                                                   const _Float16* __restrict__ Oh,
                                                   const float* __restrict__ bout,
                                                   const float* __restrict__ x,
                                                   float* __restrict__ out) {
  int w    = blockIdx.x * 8 + (threadIdx.x >> 5);
  int lane = threadIdx.x & 31;
  int nt = w & 255;
  int mt = (w >> 8) & 7;
  int b  = w >> 11;
  int lrow = lane & 15, lhi = lane >> 4;

  v8f acc = {};
  const _Float16* wbase = Wh + (size_t)(mt * 16 + lrow) * CDIM;
  const _Float16* ob    = Oh + ((size_t)b * NDIM + nt * 16 + lrow) * CDIM;
#pragma unroll
  for (int kc = 0; kc < 4; ++kc) {
    Frag16 af, bf;
    int lo = kc * 32 + (lhi ? 8 : 0);
    af.u[0] = *(const uint4*)(wbase + lo);
    af.u[1] = *(const uint4*)(wbase + lo + 16);
    int cb = kc * 32 + lhi * 8;
    const uint4* bp = (const uint4*)(ob + cb);
    bf.u[0] = bp[0]; bf.u[1] = bp[2];
    acc = wmma_f16(af, bf, acc);
  }
#pragma unroll
  for (int r = 0; r < 8; ++r) {
    int c  = mt * 16 + r + lhi * 8;
    int nn = nt * 16 + lrow;
    size_t oidx = ((size_t)b * CDIM + c) * NDIM + nn;
    out[oidx] = acc[r] + bout[c] + x[oidx];
  }
}

extern "C" void kernel_launch(void* const* d_in, const int* in_sizes, int n_in,
                              void* d_out, int out_size, void* d_ws, size_t ws_size,
                              hipStream_t stream) {
  const float* x    = (const float*)d_in[0];
  const float* wqkv = (const float*)d_in[1];
  const float* wout = (const float*)d_in[2];
  const float* bout = (const float*)d_in[3];
  float* out = (float*)d_out;

  char* ws = (char*)d_ws;
  _Float16* Qh = (_Float16*)(ws);
  _Float16* Kh = (_Float16*)(ws + (size_t)8  * 1024 * 1024);
  _Float16* Vh = (_Float16*)(ws + (size_t)16 * 1024 * 1024);
  _Float16* Oh = (_Float16*)(ws + (size_t)24 * 1024 * 1024);
  _Float16* Wh = (_Float16*)(ws + (size_t)32 * 1024 * 1024);

  wconv_kernel<<<64, 256, 0, stream>>>(wout, Wh, CDIM * CDIM);
  qkv_kernel<<<(BDIM * 384 * NDIM) / 256, 256, 0, stream>>>(x, wqkv, Qh, Kh, Vh);
  attn_kernel<<<dim3(NDIM / 128, BDIM), 256, 0, stream>>>(Qh, Kh, Vh, Oh);
  proj_kernel<<<2048, 256, 0, stream>>>(Wh, Oh, bout, x, out);
}
